// GraphColorizer_39281770889677
// MI455X (gfx1250) — matrix operand order repaired and checked by reading, not compiled, not run on hardware
//
#include <hip/hip_runtime.h>
#include <hip/hip_bf16.h>
#include <math.h>

#define EMB   512
#define NPC   256
#define NNODE 8192
#define DEG   16
#define NTHR  512

typedef __bf16 v16bf __attribute__((ext_vector_type(16)));
typedef __bf16 v8bf  __attribute__((ext_vector_type(8)));
typedef float  v8f   __attribute__((ext_vector_type(8)));

// ---------------- LDS layout (bytes) ----------------
#define SM_COLORS 0        // 8192 * 4           = 32768
#define SM_ABUF   32768    // 32*512 bf16        = 32768
#define SM_H      65536    // 18*512 f32         = 36864
#define SM_H1     102400   // 512 f32            = 2048
#define SM_H2     104448   // 448 f32 (padded)   = 1792
#define SM_LOG    106240   // 288 f32 (padded)   = 1152
#define SM_A1     107392   // 512 f32            = 2048
#define SM_A2     109440   // 512 f32            = 2048
#define SM_RED    111488   // 512 f32            = 2048
#define SM_MISC   113536   // misc               = 1024
#define SM_TOTAL  114560

static __device__ __forceinline__ unsigned short f32_to_bf16(float f) {
  union { float f; unsigned u; } x; x.f = f;
  unsigned u = x.u;
  unsigned r = u + 0x7FFFu + ((u >> 16) & 1u);         // round-to-nearest-even
  if (((u >> 23) & 0xFFu) == 0xFFu) r = u;             // inf/nan passthrough
  return (unsigned short)(r >> 16);
}

static __device__ __forceinline__ v16bf load_a_frag(const unsigned short* p) {
  // A 16x32 bf16 fragment: elems 0..15 = K(+0..15) for this lane
  union { v16bf v; v8bf h[2]; } r;
  r.h[0] = *(const v8bf*)(p);
  r.h[1] = *(const v8bf*)(p + 8);
  return r.v;
}

static __device__ __forceinline__ v8f wmma_bf16(v16bf a, v16bf b, v8f c) {
  return __builtin_amdgcn_wmma_f32_16x16x32_bf16(false, a, false, b, (short)0, c, false, false);
}

static __device__ __forceinline__ float wave_sum(float x) {
  #pragma unroll
  for (int o = 16; o > 0; o >>= 1) x += __shfl_xor(x, o, 32);
  return x;
}

static __device__ __forceinline__ float block_sum(float v, float* s_red, int tid) {
  s_red[tid] = v; __syncthreads();
  for (int st = 256; st > 0; st >>= 1) {
    if (tid < st) s_red[tid] += s_red[tid + st];
    __syncthreads();
  }
  float r = s_red[0]; __syncthreads();
  return r;
}

// Deterministic sum of squares over a global array (fixed order per thread + tree).
static __device__ float block_ssq(const float* p, int n, float* s_red, int tid) {
  float s = 0.f;
  for (int i = tid; i < n; i += NTHR) { float x = p[i]; s += x * x; }
  return block_sum(s, s_red, tid);
}

// One WMMA mat-vec stage: row 0 of Abuf (bf16, row stride 512) x B[N][K] (bf16).
static __device__ __forceinline__ void wmma_vec_stage(
    const unsigned short* Abuf, const unsigned short* Bmat, int ldb, int K,
    int ntiles, const float* bias, int nvalid, float* outv, float slope,
    int wave, int lane)
{
  const unsigned short* Ap = Abuf + (lane & 15) * EMB + ((lane >> 4) << 4);
  for (int nt = wave; nt < ntiles; nt += 16) {
    const unsigned short* Brow = Bmat + (size_t)(nt * 16 + (lane & 15)) * ldb + ((lane >> 4) << 4);
    v8f acc = {0.f,0.f,0.f,0.f,0.f,0.f,0.f,0.f};
    #pragma unroll 4
    for (int k0 = 0; k0 < K; k0 += 32) {
      v16bf b = *(const v16bf*)(Brow + k0);
      v16bf a = load_a_frag(Ap + k0);
      acc = wmma_bf16(a, b, acc);
      __builtin_prefetch(Brow + k0 + 128, 0, 1);
    }
    if (lane < 16) {               // M==0 lives in VGPR0 of lanes 0..15
      int n = nt * 16 + lane;
      if (n < nvalid) {
        float x = acc[0] + bias[n];
        outv[n] = (x > 0.f) ? x : slope * x;
      }
    }
  }
}

// ---------------- prep: zero emb, build bf16 weight layouts ----------------
__global__ void gc_prep(const float* __restrict__ W, const float* __restrict__ fc1_w,
                        const float* __restrict__ fc2_w, const float* __restrict__ fc3_w,
                        unsigned short* __restrict__ embg, unsigned short* __restrict__ W1t,
                        unsigned short* __restrict__ fc1t, unsigned short* __restrict__ fc2t,
                        unsigned short* __restrict__ fc3t)
{
  int g  = blockIdx.x * blockDim.x + threadIdx.x;
  int gs = gridDim.x * blockDim.x;
  for (int i = g; i < NNODE * EMB; i += gs) embg[i] = 0;
  for (int i = g; i < EMB * EMB; i += gs) {          // W1t[n][k] = W[k][n]
    int n = i >> 9, k = i & 511;
    W1t[i] = f32_to_bf16(W[k * EMB + n]);
  }
  for (int i = g; i < EMB * EMB; i += gs) fc1t[i] = f32_to_bf16(fc1_w[i]);   // already [N][K]
  for (int i = g; i < 400 * EMB; i += gs) fc2t[i] = f32_to_bf16(fc2_w[i]);
  for (int i = g; i < 272 * 416; i += gs) {          // zero-padded [272][416]
    int n = i / 416, k = i - n * 416;
    fc3t[i] = (n < 257 && k < 400) ? f32_to_bf16(fc3_w[n * 400 + k]) : (unsigned short)0;
  }
}

// ---------------- main: persistent sequential scan, 1 workgroup ----------------
__launch_bounds__(NTHR, 1)
__global__ void gc_main(const int* __restrict__ adj, const float* __restrict__ Wf,
                        const float* __restrict__ a_vec,
                        const float* __restrict__ fc1_w, const float* __restrict__ fc1_b,
                        const float* __restrict__ fc2_w, const float* __restrict__ fc2_b,
                        const float* __restrict__ fc3_w, const float* __restrict__ fc3_b,
                        const unsigned short* __restrict__ W1t, const unsigned short* __restrict__ fc1t,
                        const unsigned short* __restrict__ fc2t, const unsigned short* __restrict__ fc3t,
                        unsigned short* __restrict__ embg, const int* __restrict__ baseline_p,
                        float* __restrict__ out)
{
  extern __shared__ char smem[];
  int* colors           = (int*)(smem + SM_COLORS);
  unsigned short* Abuf  = (unsigned short*)(smem + SM_ABUF);
  float* hbuf           = (float*)(smem + SM_H);
  float* h1             = (float*)(smem + SM_H1);
  float* h2             = (float*)(smem + SM_H2);
  float* logits         = (float*)(smem + SM_LOG);
  float* s_a1           = (float*)(smem + SM_A1);
  float* s_a2           = (float*)(smem + SM_A2);
  float* s_red          = (float*)(smem + SM_RED);
  float* misc           = (float*)(smem + SM_MISC);
  float* s_e    = misc;            // 32
  float* s_attn = misc + 32;       // 32
  float* redv   = misc + 64;       // 16
  int*   redi   = (int*)(misc + 80);   // 16
  int*   s_nbr  = (int*)(misc + 96);   // 16
  int*   s_ucol = (int*)(misc + 112);  // 20
  float* s_scal = misc + 132;      // [0]=lpp [1]=n_used(int) [2]=maxv [3]=maxi(int) [4]=reg
  int*   s_nusedp = (int*)(s_scal + 1);
  int*   s_maxip  = (int*)(s_scal + 3);

  const int tid  = threadIdx.x;
  const int wave = tid >> 5;
  const int lane = tid & 31;

  // ---- deterministic regularizer (Frobenius norms) ----
  float r0 = block_ssq(Wf,    768 * EMB, s_red, tid);
  float r1 = block_ssq(fc1_w, EMB * EMB, s_red, tid);
  float r2 = block_ssq(fc2_w, 400 * EMB, s_red, tid);
  float r3 = block_ssq(fc3_w, 257 * 400, s_red, tid);

  // ---- init persistent state ----
  for (int i = tid; i < NNODE; i += NTHR) colors[i] = -1;
  for (int i = tid; i < EMB; i += NTHR) { s_a1[i] = a_vec[i]; s_a2[i] = a_vec[EMB + i]; }
  if (tid == 0) {
    colors[0] = 0;
    s_scal[0] = 0.f;                 // lpp
    *s_nusedp = 1;
    s_scal[4] = 0.05f * (sqrtf(r0) + sqrtf(r1) + sqrtf(r2) + sqrtf(r3));
  }
  __syncthreads();

  for (int v = 1; v < NNODE; ++v) {
    // ---- A: gather neighbors, colors, E rows (bf16) into Abuf ----
    if (tid < DEG) s_nbr[tid] = adj[(size_t)v * DEG + tid];
    __syncthreads();
    if (tid < 17) { int u = tid ? s_nbr[tid - 1] : v; s_ucol[tid] = colors[u]; }
    for (int idx = tid; idx < 17 * EMB; idx += NTHR) {
      int i = idx >> 9, c = idx & 511;
      int u = i ? s_nbr[i - 1] : v;
      Abuf[i * EMB + c] = embg[(size_t)u * EMB + c];
    }
    __syncthreads();

    // ---- B: h(17x512) = A(32x512) @ W1t via WMMA (2 m-tiles x 32 n-tiles) ----
    {
      const unsigned short* A0 = Abuf + (lane & 15) * EMB + ((lane >> 4) << 4);
      const unsigned short* A1p = A0 + 16 * EMB;
      for (int nti = 0; nti < 2; ++nti) {
        int nt = wave * 2 + nti;
        const unsigned short* Brow = W1t + (size_t)(nt * 16 + (lane & 15)) * EMB + ((lane >> 4) << 4);
        v8f acc0 = {0.f,0.f,0.f,0.f,0.f,0.f,0.f,0.f};
        v8f acc1 = {0.f,0.f,0.f,0.f,0.f,0.f,0.f,0.f};
        #pragma unroll 2
        for (int k0 = 0; k0 < EMB; k0 += 32) {
          v16bf b  = *(const v16bf*)(Brow + k0);
          v16bf a0 = load_a_frag(A0 + k0);
          v16bf a1 = load_a_frag(A1p + k0);
          acc0 = wmma_bf16(a0, b, acc0);
          acc1 = wmma_bf16(a1, b, acc1);
          __builtin_prefetch(Brow + k0 + 128, 0, 1);
        }
        int col = nt * 16 + (lane & 15);
        int rb  = (lane >> 4) << 3;
        #pragma unroll
        for (int g = 0; g < 8; ++g) {
          hbuf[(rb + g) * EMB + col] = acc0[g];
          int m1 = 16 + rb + g;
          if (m1 < 17) hbuf[m1 * EMB + col] = acc1[g];   // only row 16 is real
        }
      }
    }
    __syncthreads();

    // ---- C: one-hot contribution = gather of W rows 512..767 (f32) ----
    for (int idx = tid; idx < 17 * EMB; idx += NTHR) {
      int i = idx >> 9, c = idx & 511;
      int col = s_ucol[i];
      if (col >= 0 && col < NPC) hbuf[idx] += Wf[(size_t)(EMB + col) * EMB + c];
    }
    __syncthreads();

    // ---- D: attention scores: s2[i] = h_i . a2 ; s1 = h_0 . a1 ----
    for (int i = wave; i < 17; i += 16) {
      float p = 0.f;
      for (int c = lane; c < EMB; c += 32) p += hbuf[i * EMB + c] * s_a2[c];
      p = wave_sum(p);
      if (lane == 0) s_e[i] = p;
    }
    float s1v = block_sum(hbuf[tid] * s_a1[tid], s_red, tid);

    // ---- E: leaky(0.2) + softmax over 17 (thread 0) ----
    if (tid == 0) {
      float mx = -3.4e38f, ev[17];
      for (int i = 0; i < 17; ++i) {
        float t = s1v + s_e[i];
        t = (t > 0.f) ? t : 0.2f * t;
        ev[i] = t; if (t > mx) mx = t;
      }
      float sum = 0.f;
      for (int i = 0; i < 17; ++i) { float p = __expf(ev[i] - mx); s_attn[i] = p; sum += p; }
      float inv = 1.f / sum;
      for (int i = 0; i < 17; ++i) s_attn[i] *= inv;
    }
    __syncthreads();

    // ---- F: new_e = elu(attn @ h); write back emb[v]; stage as A row 0 ----
    {
      float acc = 0.f;
      for (int i = 0; i < 17; ++i) acc += s_attn[i] * hbuf[i * EMB + tid];
      float ne = (acc > 0.f) ? acc : (__expf(acc) - 1.f);
      unsigned short nb = f32_to_bf16(ne);
      embg[(size_t)v * EMB + tid] = nb;
      Abuf[tid] = nb;
    }
    __syncthreads();

    // ---- G: fc1 (512), leaky 0.01 ----
    wmma_vec_stage(Abuf, fc1t, EMB, EMB, 32, fc1_b, 512, h1, 0.01f, wave, lane);
    __syncthreads();
    Abuf[tid] = f32_to_bf16(h1[tid]);
    if (tid >= 400 && tid < 416) h2[tid] = 0.f;      // K padding for fc3
    __syncthreads();

    // ---- I: fc2 (400), leaky 0.01 ----
    wmma_vec_stage(Abuf, fc2t, EMB, EMB, 25, fc2_b, 400, h2, 0.01f, wave, lane);
    __syncthreads();
    if (tid < 416) Abuf[tid] = f32_to_bf16(h2[tid]);
    __syncthreads();

    // ---- K: fc3 (257 logits, K=416 padded) ----
    wmma_vec_stage(Abuf, fc3t, 416, 416, 17, fc3_b, 257, logits, 1.0f, wave, lane);
    __syncthreads();

    // ---- L: mask, softmax-max, argmax (first-index ties), update state ----
    float masked = -INFINITY;
    int   nu = *s_nusedp;
    if (tid < NPC + 1) {
      float lg = logits[tid];
      bool irrel = (tid >= nu) && (tid < NPC);
      bool amask = false;
      #pragma unroll
      for (int i = 1; i <= DEG; ++i) amask |= (s_ucol[i] == tid);
      masked = (irrel || amask) ? -INFINITY : lg;
    }
    float mv = masked;
    int   mi = (tid < NPC + 1) ? tid : 0x7fffffff;
    #pragma unroll
    for (int o = 16; o > 0; o >>= 1) {
      float ov = __shfl_xor(mv, o, 32);
      int   oi = __shfl_xor(mi, o, 32);
      if (ov > mv || (ov == mv && oi < mi)) { mv = ov; mi = oi; }
    }
    if (lane == 0) { redv[wave] = mv; redi[wave] = mi; }
    __syncthreads();
    if (tid == 0) {
      float bv = redv[0]; int bi = redi[0];
      for (int w = 1; w < 16; ++w)
        if (redv[w] > bv || (redv[w] == bv && redi[w] < bi)) { bv = redv[w]; bi = redi[w]; }
      s_scal[2] = bv; *s_maxip = bi;
    }
    __syncthreads();
    float mx = s_scal[2];
    float pex = (masked > -3.0e38f) ? __expf(masked - mx) : 0.f;
    float sum = block_sum(pex, s_red, tid);
    if (tid == 0) {
      float maxprob = 1.f / sum;                       // exp(max - max) / sum
      s_scal[0] += __logf(maxprob + 1e-8f) - __logf(1e-8f);
      int bi = *s_maxip;
      int isnew = (bi == NPC) ? 1 : 0;
      int chosen = isnew ? nu : bi;
      *s_nusedp = nu + isnew;
      colors[v] = chosen;
    }
    __syncthreads();
  }

  // ---- finalize: colors + loss ----
  for (int i = tid; i < NNODE; i += NTHR) out[i] = (float)colors[i];
  if (tid == 0) {
    float loss = ((float)(*s_nusedp) - (float)(*baseline_p)) * s_scal[0] / (float)NNODE + s_scal[4];
    out[NNODE] = loss;
  }
}

extern "C" void kernel_launch(void* const* d_in, const int* in_sizes, int n_in,
                              void* d_out, int out_size, void* d_ws, size_t ws_size,
                              hipStream_t stream) {
  const int*   adj    = (const int*)d_in[0];
  const float* W      = (const float*)d_in[1];
  const float* a      = (const float*)d_in[2];
  const float* fc1_w  = (const float*)d_in[3];
  const float* fc1_b  = (const float*)d_in[4];
  const float* fc2_w  = (const float*)d_in[5];
  const float* fc2_b  = (const float*)d_in[6];
  const float* fc3_w  = (const float*)d_in[7];
  const float* fc3_b  = (const float*)d_in[8];
  const int*   basel  = (const int*)d_in[9];
  float* out = (float*)d_out;

  char* ws = (char*)d_ws;
  unsigned short* embg = (unsigned short*)(ws + 64);           // 8192*512 bf16 (8 MB)
  unsigned short* W1t  = embg + (size_t)NNODE * EMB;           // 512*512
  unsigned short* fc1t = W1t  + (size_t)EMB * EMB;             // 512*512
  unsigned short* fc2t = fc1t + (size_t)EMB * EMB;             // 400*512
  unsigned short* fc3t = fc2t + (size_t)400 * EMB;             // 272*416 (padded)

  gc_prep<<<4096, 256, 0, stream>>>(W, fc1_w, fc2_w, fc3_w, embg, W1t, fc1t, fc2t, fc3t);
  gc_main<<<1, NTHR, SM_TOTAL, stream>>>(adj, W, a, fc1_w, fc1_b, fc2_w, fc2_b,
                                         fc3_w, fc3_b, W1t, fc1t, fc2t, fc3t,
                                         embg, basel, out);
}
